// LSTMPPO_85796266705188
// MI455X (gfx1250) — matrix operand order repaired and checked by reading, not compiled, not run on hardware
//
#include <hip/hip_runtime.h>

typedef __attribute__((ext_vector_type(16))) __bf16 v16bf;
typedef __attribute__((ext_vector_type(8)))  __bf16 v8bf;
typedef __attribute__((ext_vector_type(8)))  float  v8f;
typedef int vi4 __attribute__((vector_size(4 * sizeof(int))));

#if defined(__has_builtin)
#if __has_builtin(__builtin_amdgcn_global_load_async_to_lds_b128)
#define HAVE_ASYNC_COPY 1
#endif
#endif
#ifndef HAVE_ASYNC_COPY
#define HAVE_ASYNC_COPY 0
#endif

namespace lstmppo {

constexpr int BNV = 128;            // envs (batch)
constexpr int TST = 512;            // timesteps
constexpr int HID = 512;            // D = H = P
constexpr int NTOT = BNV * TST;     // 65536 rows
constexpr int GATES = 4 * HID;      // 2048

constexpr size_t KB = 1024, MB = 1024 * 1024;
// -------- workspace layout (~204 MB) --------
constexpr size_t WIH_PI  = 0;                  // 2 MB bf16 [2048,512]
constexpr size_t WHH_PI  = 2 * MB;
constexpr size_t WIH_VF  = 4 * MB;
constexpr size_t WHH_VF  = 6 * MB;
constexpr size_t POLW1   = 8 * MB;             // 512 KB bf16 [512,512]
constexpr size_t POLW2   = 8 * MB + 512 * KB;
constexpr size_t VALW1   = 9 * MB;
constexpr size_t VALW2   = 9 * MB + 512 * KB;
constexpr size_t ACTW    = 10 * MB;            // 32 KB bf16 [32,512]
constexpr size_t BIAS_PI = 10 * MB + 64 * KB;  // 8 KB f32 (bih+bhh)
constexpr size_t BIAS_VF = BIAS_PI + 8 * KB;
constexpr size_t H_PI    = 10 * MB + 128 * KB; // bf16 h-state (pre-masked)
constexpr size_t C_PI    = H_PI + 256 * KB;    // f32  c-state (pre-masked)
constexpr size_t H_VF    = C_PI + 256 * KB;
constexpr size_t C_VF    = H_VF + 256 * KB;
constexpr size_t BARRIER = 11 * MB + 256 * KB; // 2 monotonic counters
constexpr size_t YS_PI   = 12 * MB;            // 64 MB bf16 [N,512]
constexpr size_t YS_VF   = 76 * MB;            // 64 MB bf16 [N,512]
constexpr size_t XBF     = 140 * MB;           // 64 MB bf16 features [N,512]

__device__ inline v8f wmma_bf16(v16bf a, v16bf b, v8f c) {
  // D = A(16x32 bf16) x B(32x16 bf16) + C(16x16 f32)
  return __builtin_amdgcn_wmma_f32_16x16x32_bf16(false, a, false, b, (short)0, c,
                                                 false, false);
}

__device__ inline v16bf joinA(v8bf lo, v8bf hi) {
  v16bf r;
#pragma unroll
  for (int e = 0; e < 8; ++e) { r[e] = lo[e]; r[8 + e] = hi[e]; }
  return r;
}

__device__ inline float sigm(float x) { return 1.0f / (1.0f + __expf(-x)); }

// 16-byte global -> LDS copy; async (ASYNCcnt) when toolchain exposes it.
__device__ inline void copy16_g2l(const __bf16* g, __bf16* l) {
#if HAVE_ASYNC_COPY
  __builtin_amdgcn_global_load_async_to_lds_b128(
      (__attribute__((address_space(1))) vi4*)g,
      (__attribute__((address_space(3))) vi4*)l, 0, 0);
#else
  *(v8bf*)l = *(const v8bf*)g;
#endif
}

__device__ inline void wait_async_copies() {
#if HAVE_ASYNC_COPY
#if __has_builtin(__builtin_amdgcn_s_wait_asynccnt)
  __builtin_amdgcn_s_wait_asynccnt(0);
#else
  asm volatile("s_wait_asynccnt 0" ::: "memory");
#endif
#endif
}

// Monotonic-counter grid barrier (counters reset by prep each launch).
__device__ inline void grid_barrier(unsigned* cnt, unsigned nwg) {
  __syncthreads();
  if (threadIdx.x == 0) {
    __threadfence();
    unsigned r = atomicAdd(cnt, 1u);
    unsigned target = (r / nwg + 1u) * nwg;
    while (atomicAdd(cnt, 0u) < target) __builtin_amdgcn_s_sleep(1);
    __threadfence();
  }
  __syncthreads();
}

}  // namespace lstmppo
using namespace lstmppo;

// ---------------------------------------------------------------- prep ------
extern "C" __global__ void __launch_bounds__(256)
ppo_prep(const float* feat, const float* starts, const float* wih_pi,
         const float* whh_pi, const float* bih_pi, const float* bhh_pi,
         const float* wih_vf, const float* whh_vf, const float* bih_vf,
         const float* bhh_vf, const float* pw1, const float* pw2,
         const float* vw1, const float* vw2, const float* aw,
         const float* h0pi, const float* c0pi, const float* h0vf,
         const float* c0vf, char* ws) {
  size_t i0 = (size_t)blockIdx.x * blockDim.x + threadIdx.x;
  size_t st = (size_t)gridDim.x * blockDim.x;
  // features -> bf16 (vectorized)
  __bf16* xbf = (__bf16*)(ws + XBF);
  const float4* f4 = (const float4*)feat;
  for (size_t i = i0; i < (size_t)NTOT * HID / 4; i += st) {
    const float4 v = f4[i];
    xbf[4 * i + 0] = (__bf16)v.x;
    xbf[4 * i + 1] = (__bf16)v.y;
    xbf[4 * i + 2] = (__bf16)v.z;
    xbf[4 * i + 3] = (__bf16)v.w;
  }
  __bf16* dWihPi = (__bf16*)(ws + WIH_PI);
  __bf16* dWhhPi = (__bf16*)(ws + WHH_PI);
  __bf16* dWihVf = (__bf16*)(ws + WIH_VF);
  __bf16* dWhhVf = (__bf16*)(ws + WHH_VF);
  for (size_t i = i0; i < (size_t)GATES * HID; i += st) {
    dWihPi[i] = (__bf16)wih_pi[i];
    dWhhPi[i] = (__bf16)whh_pi[i];
    dWihVf[i] = (__bf16)wih_vf[i];
    dWhhVf[i] = (__bf16)whh_vf[i];
  }
  __bf16* dPw1 = (__bf16*)(ws + POLW1);
  __bf16* dPw2 = (__bf16*)(ws + POLW2);
  __bf16* dVw1 = (__bf16*)(ws + VALW1);
  __bf16* dVw2 = (__bf16*)(ws + VALW2);
  for (size_t i = i0; i < (size_t)HID * HID; i += st) {
    dPw1[i] = (__bf16)pw1[i];
    dPw2[i] = (__bf16)pw2[i];
    dVw1[i] = (__bf16)vw1[i];
    dVw2[i] = (__bf16)vw2[i];
  }
  __bf16* dAw = (__bf16*)(ws + ACTW);
  for (size_t i = i0; i < 32 * (size_t)HID; i += st) dAw[i] = (__bf16)aw[i];
  float* bPi = (float*)(ws + BIAS_PI);
  float* bVf = (float*)(ws + BIAS_VF);
  for (size_t i = i0; i < (size_t)GATES; i += st) {
    bPi[i] = bih_pi[i] + bhh_pi[i];
    bVf[i] = bih_vf[i] + bhh_vf[i];
  }
  // state init with the t=0 mask pre-applied (h stored bf16, c stored f32)
  __bf16* hPi = (__bf16*)(ws + H_PI);
  float* cPi = (float*)(ws + C_PI);
  __bf16* hVf = (__bf16*)(ws + H_VF);
  float* cVf = (float*)(ws + C_VF);
  for (size_t i = i0; i < (size_t)BNV * HID; i += st) {
    const int b = (int)(i >> 9);
    const float m0 = 1.0f - starts[b * TST];
    hPi[i] = (__bf16)(h0pi[i] * m0);
    cPi[i] = c0pi[i] * m0;
    hVf[i] = (__bf16)(h0vf[i] * m0);
    cVf[i] = c0vf[i] * m0;
  }
  if (i0 < 64) ((unsigned*)(ws + BARRIER))[i0] = 0u;
}

// ---------------------------------------------------------------- LSTM ------
// grid = (16, 2): x = 32-hidden-column slice owner, y = lstm (0=pi, 1=vf).
// Weights for this WG's 128 gate-columns are staged ONCE into LDS (256 KB,
// async global->LDS copy); per step, A-fragments (x_t and pre-masked h_{t-1})
// are read directly from global bf16 as contiguous b128 loads.
extern "C" __global__ void __launch_bounds__(256)
ppo_lstm(const float* __restrict__ starts, char* __restrict__ ws,
         float* __restrict__ out) {
  extern __shared__ char smem[];
  __bf16* wlds = (__bf16*)smem;            // [2][128][512] bf16: Wih | Whh
  float* sLds = (float*)(smem + 262144);   // [128] next-step mask

  const int lst = blockIdx.y;
  const __bf16* Wih = (const __bf16*)(ws + (lst ? WIH_VF : WIH_PI));
  const __bf16* Whh = (const __bf16*)(ws + (lst ? WHH_VF : WHH_PI));
  const float* bias = (const float*)(ws + (lst ? BIAS_VF : BIAS_PI));
  __bf16* Hst = (__bf16*)(ws + (lst ? H_VF : H_PI));
  float* Cst = (float*)(ws + (lst ? C_VF : C_PI));
  __bf16* ys = (__bf16*)(ws + (lst ? YS_VF : YS_PI));
  const __bf16* xbf = (const __bf16*)(ws + XBF);
  unsigned* bar = (unsigned*)(ws + BARRIER) + lst * 16;

  const int tid = threadIdx.x;
  const int lane = tid & 31;
  const int wave = tid >> 5;
  const int nsub = lane & 15;
  const int khalf = lane >> 4;
  const int hBase = blockIdx.x * 32;  // this WG's hidden-column slice
  const int rBase = wave * 16;        // this wave's 16-row strip

  // ---- one-time weight staging: LDS local row lrow = ct*16 + n holds the
  // global gate-row (ct>>1)*512 + hBase + (ct&1)*16 + n, for Wih then Whh.
  for (int e = tid * 8; e < 2 * BNV * HID; e += 256 * 8) {
    const int half = e >> 16;        // 0 = Wih, 1 = Whh
    const int le = e & 65535;
    const int lrow = le >> 9, k = le & 511;
    const int ct = lrow >> 4;
    const int grow = (ct >> 1) * HID + hBase + (ct & 1) * 16 + (lrow & 15);
    const __bf16* g = (half ? Whh : Wih) + (size_t)grow * HID + k;
    copy16_g2l(g, wlds + e);
  }
  wait_async_copies();
  __syncthreads();

  for (int t = 0; t < TST; ++t) {
    // stage next-step mask (pre-applied to h/c when written below)
    if (tid < BNV)
      sLds[tid] = (t + 1 < TST) ? (1.0f - starts[tid * TST + t + 1]) : 1.0f;

    // prefetch next step's x rows for this wave
    if (t + 1 < TST)
      __builtin_prefetch(xbf + ((size_t)((rBase + nsub) * TST + t + 1)) * HID,
                         0, 1);

    // ---- gate GEMM: g[128 x 128cols] over K = 512 (x) + 512 (h)
    const __bf16* xrow = xbf + ((size_t)((rBase + nsub) * TST + t)) * HID;
    const __bf16* hrow = Hst + (rBase + nsub) * HID;
    v8f acc[8] = {};  // 8 col-tiles: gate q = ct>>1, half = ct&1
    for (int kt = 0; kt < 32; ++kt) {
      const int kk = (kt & 15) * 32;
      const __bf16* arow = (kt < 16 ? xrow : hrow) + kk;
      v16bf a = joinA(*(const v8bf*)(arow + 16 * khalf),
                      *(const v8bf*)(arow + 8 + 16 * khalf));
      const __bf16* wbase = wlds + (kt < 16 ? 0 : BNV * HID) + kk + 16 * khalf;
#pragma unroll
      for (int ct = 0; ct < 8; ++ct) {
        v16bf b = *(const v16bf*)(wbase + (ct * 16 + nsub) * HID);
        acc[ct] = wmma_bf16(a, b, acc[ct]);
      }
    }
    grid_barrier(bar, gridDim.x);  // all WGs finished READING h state

    // ---- gates; C/D layout: M = v + 8*khalf, N = lane&15
#pragma unroll
    for (int jh = 0; jh < 2; ++jh) {
      const int col = hBase + jh * 16 + nsub;
      const float bi = bias[col];
      const float bf_ = bias[HID + col];
      const float bg = bias[2 * HID + col];
      const float bo = bias[3 * HID + col];
#pragma unroll
      for (int v = 0; v < 8; ++v) {
        const int row = rBase + khalf * 8 + v;
        const float m = sLds[row];  // mask for NEXT step (1.0 at t = T-1)
        const float gi = acc[0 + jh][v] + bi;
        const float gf = acc[2 + jh][v] + bf_;
        const float gg = acc[4 + jh][v] + bg;
        const float go = acc[6 + jh][v] + bo;
        const float cOld = Cst[row * HID + col];        // already masked
        const float cNew = sigm(gf) * cOld + sigm(gi) * tanhf(gg);
        const float hNew = sigm(go) * tanhf(cNew);
        Cst[row * HID + col] = cNew * m;                // pre-mask for t+1
        Hst[row * HID + col] = (__bf16)(hNew * m);      // pre-mask for t+1
        ys[((size_t)row * TST + t) * HID + col] = (__bf16)hNew;
        if (t == TST - 1) {  // final unmasked state straight to output
          out[(3 + 2 * lst) * NTOT + row * HID + col] = hNew;
          out[(4 + 2 * lst) * NTOT + row * HID + col] = cNew;
        }
      }
    }
    grid_barrier(bar, gridDim.x);  // state writes visible before next read
  }
}

// ---------------------------------------------------------------- head ------
// grid = (N/128, 2): x = 128-row block, y: 0 = policy branch, 1 = value.
// Fully fused MLP (L1+relu -> LDS, L2 -> LDS, then actor/critic epilogue).
extern "C" __global__ void __launch_bounds__(256)
ppo_head(const char* __restrict__ ws, const float* __restrict__ pol_b1,
         const float* __restrict__ pol_b2, const float* __restrict__ val_b1,
         const float* __restrict__ val_b2, const float* __restrict__ actor_b,
         const float* __restrict__ critic_w, const float* __restrict__ critic_b,
         float* __restrict__ out) {
  extern __shared__ char smem[];
  __bf16* H1 = (__bf16*)smem;             // [128][512] bf16
  __bf16* H2 = (__bf16*)(smem + 131072);  // [128][512] bf16
  float* red = (float*)(smem + 262144);   // [128][32] f32 logits

  const int branch = blockIdx.y;
  const int R = blockIdx.x * 128;
  const __bf16* ysb = (const __bf16*)(ws + (branch ? YS_VF : YS_PI));
  const __bf16* W1 = (const __bf16*)(ws + (branch ? VALW1 : POLW1));
  const __bf16* W2 = (const __bf16*)(ws + (branch ? VALW2 : POLW2));
  const __bf16* AW = (const __bf16*)(ws + ACTW);
  const float* b1 = branch ? val_b1 : pol_b1;
  const float* b2 = branch ? val_b2 : pol_b2;

  const int tid = threadIdx.x, lane = tid & 31, wave = tid >> 5;
  const int nsub = lane & 15, khalf = lane >> 4;
  const int rBase = wave * 16;

  // layer 1: H1 = relu(ys @ W1^T + b1)
  const __bf16* Abase = ysb + (size_t)(R + rBase + nsub) * HID;
  for (int grp = 0; grp < 4; ++grp) {
    v8f acc[8] = {};
    for (int kt = 0; kt < 16; ++kt) {
      const int kk = kt * 32;
      v16bf a = joinA(*(const v8bf*)(Abase + kk + 16 * khalf),
                      *(const v8bf*)(Abase + kk + 8 + 16 * khalf));
#pragma unroll
      for (int ct = 0; ct < 8; ++ct) {
        const int wrow = grp * 128 + ct * 16 + nsub;
        v16bf b = *(const v16bf*)(W1 + (size_t)wrow * HID + kk + 16 * khalf);
        acc[ct] = wmma_bf16(a, b, acc[ct]);
      }
    }
#pragma unroll
    for (int ct = 0; ct < 8; ++ct) {
      const int col = grp * 128 + ct * 16 + nsub;
      const float bb = b1[col];
#pragma unroll
      for (int v = 0; v < 8; ++v) {
        const int row = rBase + khalf * 8 + v;
        const float x = acc[ct][v] + bb;
        H1[row * HID + col] = (__bf16)(x > 0.0f ? x : 0.0f);
      }
    }
  }
  __syncthreads();
  // layer 2: H2 = H1 @ W2^T + b2
  const __bf16* A2 = H1 + (rBase + nsub) * HID;
  for (int grp = 0; grp < 4; ++grp) {
    v8f acc[8] = {};
    for (int kt = 0; kt < 16; ++kt) {
      const int kk = kt * 32;
      v16bf a = joinA(*(const v8bf*)(A2 + kk + 16 * khalf),
                      *(const v8bf*)(A2 + kk + 8 + 16 * khalf));
#pragma unroll
      for (int ct = 0; ct < 8; ++ct) {
        const int wrow = grp * 128 + ct * 16 + nsub;
        v16bf b = *(const v16bf*)(W2 + (size_t)wrow * HID + kk + 16 * khalf);
        acc[ct] = wmma_bf16(a, b, acc[ct]);
      }
    }
#pragma unroll
    for (int ct = 0; ct < 8; ++ct) {
      const int col = grp * 128 + ct * 16 + nsub;
      const float bb = b2[col];
#pragma unroll
      for (int v = 0; v < 8; ++v) {
        const int row = rBase + khalf * 8 + v;
        H2[row * HID + col] = (__bf16)(acc[ct][v] + bb);
      }
    }
  }
  __syncthreads();

  if (branch == 0) {
    // logits [128,32] = H2 @ actor_w^T + actor_b, then argmax/log-softmax
    v8f acc[2] = {};
    const __bf16* A3 = H2 + (rBase + nsub) * HID;
    for (int kt = 0; kt < 16; ++kt) {
      const int kk = kt * 32;
      v16bf a = joinA(*(const v8bf*)(A3 + kk + 16 * khalf),
                      *(const v8bf*)(A3 + kk + 8 + 16 * khalf));
#pragma unroll
      for (int ct = 0; ct < 2; ++ct) {
        const int wrow = ct * 16 + nsub;
        v16bf b = *(const v16bf*)(AW + (size_t)wrow * HID + kk + 16 * khalf);
        acc[ct] = wmma_bf16(a, b, acc[ct]);
      }
    }
#pragma unroll
    for (int ct = 0; ct < 2; ++ct) {
      const int col = ct * 16 + nsub;
      const float bb = actor_b[col];
#pragma unroll
      for (int v = 0; v < 8; ++v) {
        const int row = rBase + khalf * 8 + v;
        red[row * 32 + col] = acc[ct][v] + bb;
      }
    }
    __syncthreads();
    if (tid < 128) {
      float mx = -1e30f;
      int am = 0;
      for (int j = 0; j < 32; ++j) {
        const float l = red[tid * 32 + j];
        if (l > mx) { mx = l; am = j; }
      }
      float s = 0.0f;
      for (int j = 0; j < 32; ++j) s += __expf(red[tid * 32 + j] - mx);
      const float lse = mx + __logf(s);
      out[R + tid] = (float)am;            // actions (as float)
      out[2 * NTOT + R + tid] = mx - lse;  // log_prob of argmax action
    }
  } else {
    if (tid < 128) {
      float a = 0.0f;
      for (int k = 0; k < HID; ++k) a += (float)H2[tid * HID + k] * critic_w[k];
      out[NTOT + R + tid] = a + critic_b[0];  // values
    }
  }
}

// --------------------------------------------------------------- launch -----
extern "C" void kernel_launch(void* const* d_in, const int* in_sizes, int n_in,
                              void* d_out, int out_size, void* d_ws,
                              size_t ws_size, hipStream_t stream) {
  (void)in_sizes; (void)n_in; (void)out_size; (void)ws_size;
  const float* features = (const float*)d_in[0];
  const float* starts = (const float*)d_in[1];
  const float* h0_pi = (const float*)d_in[2];
  const float* c0_pi = (const float*)d_in[3];
  const float* h0_vf = (const float*)d_in[4];
  const float* c0_vf = (const float*)d_in[5];
  const float* wih_pi = (const float*)d_in[6];
  const float* whh_pi = (const float*)d_in[7];
  const float* bih_pi = (const float*)d_in[8];
  const float* bhh_pi = (const float*)d_in[9];
  const float* wih_vf = (const float*)d_in[10];
  const float* whh_vf = (const float*)d_in[11];
  const float* bih_vf = (const float*)d_in[12];
  const float* bhh_vf = (const float*)d_in[13];
  const float* pol_w1 = (const float*)d_in[14];
  const float* pol_b1 = (const float*)d_in[15];
  const float* pol_w2 = (const float*)d_in[16];
  const float* pol_b2 = (const float*)d_in[17];
  const float* val_w1 = (const float*)d_in[18];
  const float* val_b1 = (const float*)d_in[19];
  const float* val_w2 = (const float*)d_in[20];
  const float* val_b2 = (const float*)d_in[21];
  const float* actor_w = (const float*)d_in[22];
  const float* actor_b = (const float*)d_in[23];
  const float* critic_w = (const float*)d_in[24];
  const float* critic_b = (const float*)d_in[25];
  char* ws = (char*)d_ws;
  float* out = (float*)d_out;

  ppo_prep<<<1024, 256, 0, stream>>>(features, starts, wih_pi, whh_pi, bih_pi,
                                     bhh_pi, wih_vf, whh_vf, bih_vf, bhh_vf,
                                     pol_w1, pol_w2, val_w1, val_w2, actor_w,
                                     h0_pi, c0_pi, h0_vf, c0_vf, ws);
  const size_t lds_lstm = 2 * 128 * 512 * 2 + 512;  // 256 KB weights + mask
  ppo_lstm<<<dim3(16, 2), 256, lds_lstm, stream>>>(starts, ws, out);
  const size_t lds_head = 2 * 128 * 512 * 2 + 128 * 32 * 4;  // 272 KB
  ppo_head<<<dim3(512, 2), 256, lds_head, stream>>>(ws, pol_b1, pol_b2, val_b1,
                                                    val_b2, actor_b, critic_w,
                                                    critic_b, out);
}
